// ButterflyGlobalLinear_23029614641364
// MI455X (gfx1250) — matrix operand order repaired and checked by reading, not compiled, not run on hardware
//
#include <hip/hip_runtime.h>

#define M_ROWS 16384
#define N_COLS 2048
#define K_DIM  2048

#define BM 128
#define BN 128
#define BK 32
#define LDA 40   // padded halves per LDS row (80B): conflict-free 16B frag reads

typedef __attribute__((ext_vector_type(16))) __bf16 v16bf;
typedef __attribute__((ext_vector_type(8)))  __bf16 v8bf;
typedef __attribute__((ext_vector_type(8)))  float  v8f;
typedef int v4i __attribute__((vector_size(16)));   // matches builtin param type

#if defined(__gfx1250__) && __has_builtin(__builtin_amdgcn_global_load_async_to_lds_b128) && __has_builtin(__builtin_amdgcn_s_wait_asynccnt)
#define ASYNC_OK 1
#else
#define ASYNC_OK 0
#endif

// ---------------- bf16 convert (round-to-nearest-even) ----------------
__device__ __forceinline__ unsigned short f2bf(float f) {
  unsigned int u = __float_as_uint(f);
  u += 0x7FFFu + ((u >> 16) & 1u);
  return (unsigned short)(u >> 16);
}

__device__ __forceinline__ void pack8(unsigned short* dst,
                                      float4 a, float4 b) {
  union { unsigned short s[8]; uint4 v; } u;
  u.s[0] = f2bf(a.x); u.s[1] = f2bf(a.y); u.s[2] = f2bf(a.z); u.s[3] = f2bf(a.w);
  u.s[4] = f2bf(b.x); u.s[5] = f2bf(b.y); u.s[6] = f2bf(b.z); u.s[7] = f2bf(b.w);
  *(uint4*)dst = u.v;
}

// Wb[o*K + i] = bf16(mask*weight)
__global__ void k_prep_w(const float* __restrict__ w, const float* __restrict__ m,
                         unsigned short* __restrict__ out) {
  size_t i = ((size_t)blockIdx.x * blockDim.x + threadIdx.x) * 8;
  float4 a0 = *(const float4*)(w + i);
  float4 a1 = *(const float4*)(w + i + 4);
  float4 m0 = *(const float4*)(m + i);
  float4 m1 = *(const float4*)(m + i + 4);
  a0.x *= m0.x; a0.y *= m0.y; a0.z *= m0.z; a0.w *= m0.w;
  a1.x *= m1.x; a1.y *= m1.y; a1.z *= m1.z; a1.w *= m1.w;
  pack8(out + i, a0, a1);
}

// Xb = bf16(x)
__global__ void k_prep_x(const float* __restrict__ x, unsigned short* __restrict__ out) {
  size_t i = ((size_t)blockIdx.x * blockDim.x + threadIdx.x) * 8;
  float4 a0 = *(const float4*)(x + i);
  float4 a1 = *(const float4*)(x + i + 4);
  pack8(out + i, a0, a1);
}

// ---------------- 16B global -> LDS copy (async if available) ----------------
__device__ __forceinline__ void cp16(__bf16* lds_dst, const unsigned short* gsrc) {
#if ASYNC_OK
  __builtin_amdgcn_global_load_async_to_lds_b128((v4i*)gsrc, (v4i*)lds_dst, 0, 0);
#else
  *(uint4*)lds_dst = *(const uint4*)gsrc;
#endif
}

// ---------------- bf16 WMMA GEMM:  out[M][N] = Xb[M][K] * Wb[N][K]^T + bias ----------------
__global__ __launch_bounds__(256)
void k_gemm_bf16(const unsigned short* __restrict__ xb,
                 const unsigned short* __restrict__ wb,
                 const float* __restrict__ bias,
                 float* __restrict__ out) {
  __shared__ __attribute__((aligned(16))) __bf16 As[2][BM * LDA];
  __shared__ __attribute__((aligned(16))) __bf16 Bs[2][BN * LDA];

  const int t    = threadIdx.x;
  const int lane = t & 31;
  const int wid  = t >> 5;
  const int wm   = wid & 3;   // 4 wave-rows  of 32 M
  const int wn   = wid >> 2;  // 2 wave-cols  of 64 N

  const int m0 = blockIdx.y * BM;
  const int n0 = blockIdx.x * BN;

  // staging: 512 chunks of 16B per tile, 2 per thread
  const int srow = t >> 2;           // 0..63
  const int sc8  = (t & 3) * 8;      // half-offset 0/8/16/24

  const v8f vzero = {0.f, 0.f, 0.f, 0.f, 0.f, 0.f, 0.f, 0.f};
  v8f acc[2][4];
#pragma unroll
  for (int mi = 0; mi < 2; ++mi)
#pragma unroll
    for (int ni = 0; ni < 4; ++ni) acc[mi][ni] = vzero;

  auto stage = [&](int buf, int k0) {
    const unsigned short* ga = xb + (size_t)(m0 + srow) * K_DIM + k0 + sc8;
    cp16(&As[buf][srow * LDA + sc8], ga);
    cp16(&As[buf][(srow + 64) * LDA + sc8], ga + (size_t)64 * K_DIM);
    const unsigned short* gb = wb + (size_t)(n0 + srow) * K_DIM + k0 + sc8;
    cp16(&Bs[buf][srow * LDA + sc8], gb);
    cp16(&Bs[buf][(srow + 64) * LDA + sc8], gb + (size_t)64 * K_DIM);
  };

  stage(0, 0);
#if ASYNC_OK
  __builtin_amdgcn_s_wait_asynccnt(0);
#endif
  __syncthreads();

  // per-lane fragment addressing (ISA 7.12.2, wave32)
  const int fr  = (lane & 15) * LDA;         // row within 16-row tile
  const int aK  = (lane < 16) ? 0 : 8;       // A: halves {aK..aK+7, aK+16..aK+23}
  const int bK  = (lane < 16) ? 0 : 8;       // B: halves {bK..bK+7, bK+16..bK+23}

  for (int kt = 0; kt < K_DIM / BK; ++kt) {
    const int cur = kt & 1;
    if (kt + 1 < K_DIM / BK) stage(cur ^ 1, (kt + 1) * BK);

    const __bf16* abuf = As[cur];
    const __bf16* bbuf = Bs[cur];

    v16bf afrag[2];
#pragma unroll
    for (int mi = 0; mi < 2; ++mi) {
      const __bf16* p = abuf + (wm * 32 + mi * 16) * LDA + fr;
      v8bf lo = *(const v8bf*)(p + aK);
      v8bf hi = *(const v8bf*)(p + aK + 16);
      afrag[mi] = __builtin_shufflevector(lo, hi, 0, 1, 2, 3, 4, 5, 6, 7,
                                          8, 9, 10, 11, 12, 13, 14, 15);
    }
#pragma unroll
    for (int ni = 0; ni < 4; ++ni) {
      const __bf16* p = bbuf + (wn * 64 + ni * 16) * LDA + fr + bK;
      v8bf lo = *(const v8bf*)(p);
      v8bf hi = *(const v8bf*)(p + 16);
      v16bf bfrag = __builtin_shufflevector(lo, hi, 0, 1, 2, 3, 4, 5, 6, 7,
                                            8, 9, 10, 11, 12, 13, 14, 15);
#pragma unroll
      for (int mi = 0; mi < 2; ++mi) {
        acc[mi][ni] = __builtin_amdgcn_wmma_f32_16x16x32_bf16(
            false, afrag[mi], false, bfrag, (short)0, acc[mi][ni], false, false);
      }
    }

#if ASYNC_OK
    __builtin_amdgcn_s_wait_asynccnt(0);
#endif
    __syncthreads();
  }

  // epilogue: C/D layout — VGPR r, lanes 0-15: M=r, lanes 16-31: M=r+8; N=lane%16
  const int col    = lane & 15;
  const int rowoff = (lane >> 4) * 8;
#pragma unroll
  for (int mi = 0; mi < 2; ++mi) {
#pragma unroll
    for (int ni = 0; ni < 4; ++ni) {
      const int n = n0 + wn * 64 + ni * 16 + col;
      const float b = bias[n];
      float* op = out + (size_t)(m0 + wm * 32 + mi * 16 + rowoff) * N_COLS + n;
#pragma unroll
      for (int r = 0; r < 8; ++r) op[(size_t)r * N_COLS] = acc[mi][ni][r] + b;
    }
  }
}

extern "C" void kernel_launch(void* const* d_in, const int* in_sizes, int n_in,
                              void* d_out, int out_size, void* d_ws, size_t ws_size,
                              hipStream_t stream) {
  const float* x    = (const float*)d_in[0];
  const float* mask = (const float*)d_in[1];
  const float* w    = (const float*)d_in[2];
  const float* bias = (const float*)d_in[3];

  unsigned short* wb = (unsigned short*)d_ws;                 //  8 MB
  unsigned short* xb = wb + (size_t)N_COLS * K_DIM;           // 64 MB

  const int wBlocks = (N_COLS * K_DIM) / (8 * 256);           // 2048
  const int xBlocks = (M_ROWS * K_DIM) / (8 * 256);           // 16384
  k_prep_w<<<wBlocks, 256, 0, stream>>>(w, mask, wb);
  k_prep_x<<<xBlocks, 256, 0, stream>>>(x, xb);

  dim3 grid(N_COLS / BN, M_ROWS / BM);                        // 16 x 128
  k_gemm_bf16<<<grid, 256, 0, stream>>>(xb, wb, bias, (float*)d_out);
}
